// ContactMapGeneration_10539849744807
// MI455X (gfx1250) — matrix operand order repaired and checked by reading, not compiled, not run on hardware
//
#include <hip/hip_runtime.h>
#include <hip/hip_bf16.h>

typedef __attribute__((ext_vector_type(16))) _Float16 v16h;
typedef __attribute__((ext_vector_type(8)))  float    v8f;
typedef __attribute__((ext_vector_type(2)))  _Float16 h2;
typedef __attribute__((ext_vector_type(8)))  _Float16 h8;

#define Bc  16
#define Nc  65536
#define NPc 2048
#define Mc  (Bc * NPc)   // 32768 sampled points total

// ---------------------------------------------------------------------------
// AoS -> SoA transpose of mesh coordinates (coalesced FPS reads, L2 resident)
// ---------------------------------------------------------------------------
__global__ void soa_kernel(const float* __restrict__ mesh,
                           float* __restrict__ xs, float* __restrict__ ys,
                           float* __restrict__ zs) {
    int g = blockIdx.x * blockDim.x + threadIdx.x;
    if (g >= Bc * Nc) return;
    xs[g] = mesh[g * 3 + 0];
    ys[g] = mesh[g * 3 + 1];
    zs[g] = mesh[g * 3 + 2];
}

// ---------------------------------------------------------------------------
// Per-batch centroid + object scale (max distance from centroid)
// ---------------------------------------------------------------------------
__global__ void scale_kernel(const float* __restrict__ xs,
                             const float* __restrict__ ys,
                             const float* __restrict__ zs,
                             float* __restrict__ scale_out) {
    __shared__ float red[256];
    const int b = blockIdx.x, tid = threadIdx.x;
    float sx = 0.f, sy = 0.f, sz = 0.f;
    for (int n = tid; n < Nc; n += 256) {
        sx += xs[b * Nc + n]; sy += ys[b * Nc + n]; sz += zs[b * Nc + n];
    }
    float cx, cy, cz;
    red[tid] = sx; __syncthreads();
    for (int s = 128; s > 0; s >>= 1) { if (tid < s) red[tid] += red[tid + s]; __syncthreads(); }
    cx = red[0] / (float)Nc; __syncthreads();
    red[tid] = sy; __syncthreads();
    for (int s = 128; s > 0; s >>= 1) { if (tid < s) red[tid] += red[tid + s]; __syncthreads(); }
    cy = red[0] / (float)Nc; __syncthreads();
    red[tid] = sz; __syncthreads();
    for (int s = 128; s > 0; s >>= 1) { if (tid < s) red[tid] += red[tid + s]; __syncthreads(); }
    cz = red[0] / (float)Nc; __syncthreads();

    float mx = 0.f;
    for (int n = tid; n < Nc; n += 256) {
        float dx = xs[b * Nc + n] - cx, dy = ys[b * Nc + n] - cy, dz = zs[b * Nc + n] - cz;
        mx = fmaxf(mx, dx * dx + dy * dy + dz * dz);
    }
    red[tid] = mx; __syncthreads();
    for (int s = 128; s > 0; s >>= 1) { if (tid < s) red[tid] = fmaxf(red[tid], red[tid + s]); __syncthreads(); }
    if (tid == 0) scale_out[b] = sqrtf(red[0]);
}

// ---------------------------------------------------------------------------
// Farthest point sampling: one workgroup per batch, distance[] lives in LDS
// (256 KB of the 320 KB CDNA5 WGP LDS). 2048 serial argmax rounds.
// ---------------------------------------------------------------------------
__global__ void fps_kernel(const float* __restrict__ xs,
                           const float* __restrict__ ys,
                           const float* __restrict__ zs,
                           const int* __restrict__ farthest_init,
                           int* __restrict__ idx_out) {
    extern __shared__ unsigned char smem[];
    float* dist = (float*)smem;                       // Nc floats
    float* redv = (float*)(smem + (size_t)Nc * 4);    // 1024 floats
    int*   redi = (int*)(smem + (size_t)Nc * 4 + 4096);
    int*   bcast = (int*)(smem + (size_t)Nc * 4 + 8192);
    const int b = blockIdx.x, tid = threadIdx.x;

    for (int n = tid; n < Nc; n += 1024) dist[n] = 1e10f;
    int farthest = farthest_init[b];
    __syncthreads();

    for (int i = 0; i < NPc; ++i) {
        if (tid == 0) idx_out[b * NPc + i] = farthest;
        const float cx = xs[b * Nc + farthest];
        const float cy = ys[b * Nc + farthest];
        const float cz = zs[b * Nc + farthest];
        float bestv = -1.f; int besti = 0;
        for (int n = tid; n < Nc; n += 1024) {
            float dx = xs[b * Nc + n] - cx, dy = ys[b * Nc + n] - cy, dz = zs[b * Nc + n] - cz;
            float d = dx * dx + dy * dy + dz * dz;
            float dm = fminf(dist[n], d);
            dist[n] = dm;
            if (dm > bestv) { bestv = dm; besti = n; }
        }
        redv[tid] = bestv; redi[tid] = besti;
        __syncthreads();
        for (int s = 512; s > 0; s >>= 1) {
            if (tid < s) {
                float v2 = redv[tid + s]; int i2 = redi[tid + s];
                if (v2 > redv[tid] || (v2 == redv[tid] && i2 < redi[tid])) {
                    redv[tid] = v2; redi[tid] = i2;
                }
            }
            __syncthreads();
        }
        if (tid == 0) bcast[0] = redi[0];
        __syncthreads();
        farthest = bcast[0];
        __syncthreads();
    }
}

// ---------------------------------------------------------------------------
// Gather sampled points, normalize by scale, build encoder input [cm, pcn]
// ---------------------------------------------------------------------------
__global__ void gather_kernel(const float* __restrict__ xs, const float* __restrict__ ys,
                              const float* __restrict__ zs, const float* __restrict__ cmap,
                              const int* __restrict__ idx, const float* __restrict__ scale,
                              float* __restrict__ pcn, float* __restrict__ enc_in) {
    int p = blockIdx.x * blockDim.x + threadIdx.x;
    if (p >= Mc) return;
    int b = p / NPc;
    int i = idx[p];
    float inv = 1.f / scale[b];
    float px = xs[b * Nc + i] * inv, py = ys[b * Nc + i] * inv, pz = zs[b * Nc + i] * inv;
    pcn[p * 3 + 0] = px; pcn[p * 3 + 1] = py; pcn[p * 3 + 2] = pz;
    enc_in[p * 4 + 0] = cmap[(size_t)b * Nc + i];
    enc_in[p * 4 + 1] = px; enc_in[p * 4 + 2] = py; enc_in[p * 4 + 3] = pz;
}

// ---------------------------------------------------------------------------
// Tiny-K first MLP layer: out = relu(in[*,K] @ W[K,64] + b), store f16
// ---------------------------------------------------------------------------
template<int K>
__global__ void mlp1_kernel(const float* __restrict__ in, const float* __restrict__ W,
                            const float* __restrict__ bias, _Float16* __restrict__ out) {
    int g = blockIdx.x * blockDim.x + threadIdx.x;
    if (g >= Mc * 64) return;
    int p = g >> 6, o = g & 63;
    float s = bias[o];
#pragma unroll
    for (int k = 0; k < K; ++k) s += in[p * K + k] * W[k * 64 + o];
    out[g] = (_Float16)fmaxf(s, 0.f);
}

__global__ void convert_f16_kernel(const float* __restrict__ src, _Float16* __restrict__ dst, int count) {
    int g = blockIdx.x * blockDim.x + threadIdx.x;
    if (g < count) dst[g] = (_Float16)src[g];
}

__global__ void zero_f32_kernel(float* __restrict__ p, int count) {
    int g = blockIdx.x * blockDim.x + threadIdx.x;
    if (g < count) p[g] = 0.f;
}

__global__ void copy_f32_kernel(const float* __restrict__ src, float* __restrict__ dst, int count) {
    int g = blockIdx.x * blockDim.x + threadIdx.x;
    if (g < count) dst[g] = src[g];
}

// ---------------------------------------------------------------------------
// WMMA f16 GEMM: C[M,Nn] = act(A[M,K] @ W[K,Nn] + bias/rowvec)
// Workgroup tile 64x64, 8 waves. Each wave owns a 16x32 slab: one A fragment
// reused across two B fragments -> 2x v_wmma_f32_16x16x32_f16 per K-step.
// LDS tiles are double-buffered; next tile's global loads are issued into
// registers before the current tile's WMMAs, so only ONE barrier per K-step.
// ACT: 0 none, 1 relu, 2 leaky(0.2). POOL: 1 -> atomicMax into pool[b,Nn].
// ---------------------------------------------------------------------------
template<int ACT, int POOL>
__global__ __launch_bounds__(256)
void wmma_gemm_kernel(const _Float16* __restrict__ A, const _Float16* __restrict__ W,
                      const float* __restrict__ bias, const float* __restrict__ rowvec,
                      _Float16* __restrict__ Cout, float* __restrict__ pool,
                      int K, int Nn) {
    const int n0 = blockIdx.x * 64;
    const int m0 = blockIdx.y * 64;
    const int tid = threadIdx.x;
    const int wave = tid >> 5, lane = tid & 31;
    const int mt = (wave & 3) * 16;          // wave's 16-row slab
    const int ntb = (wave >> 2) * 32;        // wave's 32-col slab (2 x 16)
    const int lh = lane & 15, hi = lane >> 4;

    __shared__ __align__(16) _Float16 Alds[2][64 * 40];
    __shared__ __align__(16) _Float16 Blds[2][64 * 40];  // transposed [n][k]

    // staging coordinates (16B per thread for both A and B)
    const int ar = tid >> 2, ac8 = (tid & 3) * 8;        // A: row, k-chunk
    const int bk = tid >> 3, bn8 = (tid & 7) * 8;        // B: k-row, n-chunk

    v8f c0 = {0.f, 0.f, 0.f, 0.f, 0.f, 0.f, 0.f, 0.f};
    v8f c1 = {0.f, 0.f, 0.f, 0.f, 0.f, 0.f, 0.f, 0.f};

    // prologue: stage first K-tile into buffer 0
    {
        h8 av = *(const h8*)&A[(size_t)(m0 + ar) * K + ac8];
        h8 bv = *(const h8*)&W[(size_t)bk * Nn + n0 + bn8];
        *(h8*)&Alds[0][ar * 40 + ac8] = av;
#pragma unroll
        for (int q = 0; q < 8; ++q) Blds[0][(bn8 + q) * 40 + bk] = bv[q];
    }

    int cur = 0;
    for (int k0 = 0; k0 < K; k0 += 32) {
        __syncthreads();   // buffer `cur` is ready for all waves

        // issue next tile's global loads early (overlap with WMMA issue below)
        const bool nxt = (k0 + 32) < K;
        h8 av, bv;
        if (nxt) {
            av = *(const h8*)&A[(size_t)(m0 + ar) * K + (k0 + 32) + ac8];
            bv = *(const h8*)&W[(size_t)(k0 + 32 + bk) * Nn + n0 + bn8];
        }

        // build fragments from buffer `cur`
        v16h a, b0, b1;
#pragma unroll
        for (int j = 0; j < 8; ++j) {
            // A 16x32 f16 fragment: lanes 0-15 hold K{0..7,16..23}, 16-31 K{8..15,24..31}
            int ka = ((j < 4) ? (2 * j) : (16 + 2 * (j - 4))) + 8 * hi;
            h2 pa = *(const h2*)&Alds[cur][(mt + lh) * 40 + ka];
            a[2 * j] = pa.x; a[2 * j + 1] = pa.y;
            // B 32x16 fragment: lanes 0-15 K{0..7,16..23}, 16-31 K{8..15,24..31}, N=lane&15
            int kb = ((j < 4) ? (2 * j) : (16 + 2 * (j - 4))) + 8 * hi;
            h2 p0 = *(const h2*)&Blds[cur][(ntb + lh) * 40 + kb];
            h2 p1 = *(const h2*)&Blds[cur][(ntb + 16 + lh) * 40 + kb];
            b0[2 * j] = p0.x; b0[2 * j + 1] = p0.y;
            b1[2 * j] = p1.x; b1[2 * j + 1] = p1.y;
        }
        c0 = __builtin_amdgcn_wmma_f32_16x16x32_f16(false, a, false, b0, (short)0, c0, false, false);
        c1 = __builtin_amdgcn_wmma_f32_16x16x32_f16(false, a, false, b1, (short)0, c1, false, false);

        // write next tile into the other buffer (no barrier needed here;
        // next iteration's top barrier publishes it)
        if (nxt) {
            *(h8*)&Alds[cur ^ 1][ar * 40 + ac8] = av;
#pragma unroll
            for (int q = 0; q < 8; ++q) Blds[cur ^ 1][(bn8 + q) * 40 + bk] = bv[q];
        }
        cur ^= 1;
    }

    // Epilogue: C/D layout — VGPR r: lanes 0-15 -> M=r, lanes 16-31 -> M=8+r
#pragma unroll
    for (int r = 0; r < 8; ++r) {
        const int m = m0 + mt + r + 8 * hi;
        const int bidx = m / NPc;
#pragma unroll
        for (int t = 0; t < 2; ++t) {
            const int n = n0 + ntb + t * 16 + lh;
            float v = t ? c1[r] : c0[r];
            v += rowvec ? rowvec[bidx * Nn + n] : bias[n];
            if (ACT == 1) v = fmaxf(v, 0.f);
            else if (ACT == 2) v = v > 0.f ? v : 0.2f * v;
            if (POOL) {
                // post-relu values are >= 0 -> int-bit max == float max
                atomicMax((int*)pool + bidx * Nn + n, __float_as_int(v));
            } else {
                Cout[(size_t)m * Nn + n] = (_Float16)v;
            }
        }
    }
}

// ---------------------------------------------------------------------------
// Small dense layers over B=16 rows (VAE encoder)
// ---------------------------------------------------------------------------
__global__ void gemv_kernel(const float* __restrict__ in, const float* __restrict__ W,
                            const float* __restrict__ bias, float* __restrict__ out,
                            int K, int Nn) {
    int g = blockIdx.x * blockDim.x + threadIdx.x;
    if (g >= Bc * Nn) return;
    int b = g / Nn, n = g % Nn;
    float s = bias[n];
    for (int k = 0; k < K; ++k) s += in[b * K + k] * W[k * Nn + n];
    out[g] = s;
}

__global__ void bn_relu_kernel(float* __restrict__ x, const float* __restrict__ g,
                               const float* __restrict__ beta, int Nn) {
    int n = blockIdx.x * blockDim.x + threadIdx.x;
    if (n >= Nn) return;
    float m = 0.f;
    for (int b = 0; b < Bc; ++b) m += x[b * Nn + n];
    m *= (1.f / Bc);
    float v = 0.f;
    for (int b = 0; b < Bc; ++b) { float d = x[b * Nn + n] - m; v += d * d; }
    v *= (1.f / Bc);
    float inv = rsqrtf(v + 1e-5f);
    for (int b = 0; b < Bc; ++b) {
        float y = g[n] * (x[b * Nn + n] - m) * inv + beta[n];
        x[b * Nn + n] = fmaxf(y, 0.f);
    }
}

__global__ void cvec64_kernel(const float* __restrict__ hv, const float* __restrict__ eps,
                              float* __restrict__ cv64) {
    int g = blockIdx.x * blockDim.x + threadIdx.x;
    if (g >= Bc * 64) return;
    int b = g / 64, j = g % 64;
    cv64[g] = hv[b * 128 + j] + eps[g] * expf(0.5f * hv[b * 128 + 64 + j]);
}

// Fold per-batch broadcast rows of dec_w1 into a per-batch constant vector:
// cvec[b,n] = gfeat@W[0:1024] + text@W[1088:1600] + scale*W[1600] + cv64@W[1601:1665] + b1
__global__ void cvec_kernel(const float* __restrict__ gfeat, const float* __restrict__ text,
                            const float* __restrict__ scale, const float* __restrict__ cv64,
                            const float* __restrict__ dw1, const float* __restrict__ db1,
                            float* __restrict__ cvec) {
    int g = blockIdx.x * blockDim.x + threadIdx.x;
    if (g >= Bc * 512) return;
    int b = g / 512, n = g % 512;
    float s = db1[n];
    for (int k = 0; k < 1024; ++k) s += gfeat[b * 1024 + k] * dw1[k * 512 + n];
    for (int t = 0; t < 512; ++t)  s += text[b * 512 + t] * dw1[(1088 + t) * 512 + n];
    s += scale[b] * dw1[1600 * 512 + n];
    for (int j = 0; j < 64; ++j)   s += cv64[b * 64 + j] * dw1[(1601 + j) * 512 + n];
    cvec[g] = s;
}

__global__ void refine_kernel(const _Float16* __restrict__ h3, const float* __restrict__ w4,
                              const float* __restrict__ b4, float* __restrict__ out) {
    int p = blockIdx.x * blockDim.x + threadIdx.x;
    if (p >= Mc) return;
    float s = b4[0];
#pragma unroll 8
    for (int j = 0; j < 128; ++j) s += (float)h3[(size_t)p * 128 + j] * w4[j];
    out[p] = s;
}

// ---------------------------------------------------------------------------
extern "C" void kernel_launch(void* const* d_in, const int* in_sizes, int n_in,
                              void* d_out, int out_size, void* d_ws, size_t ws_size,
                              hipStream_t stream) {
    const float* mesh   = (const float*)d_in[0];
    const float* cmap   = (const float*)d_in[1];
    const float* text   = (const float*)d_in[2];
    const float* eps    = (const float*)d_in[3];
    const float* pn3_w1 = (const float*)d_in[4];  const float* pn3_b1 = (const float*)d_in[5];
    const float* pn3_w2 = (const float*)d_in[6];  const float* pn3_b2 = (const float*)d_in[7];
    const float* pn3_w3 = (const float*)d_in[8];  const float* pn3_b3 = (const float*)d_in[9];
    const float* pn4_w1 = (const float*)d_in[10]; const float* pn4_b1 = (const float*)d_in[11];
    const float* pn4_w2 = (const float*)d_in[12]; const float* pn4_b2 = (const float*)d_in[13];
    const float* pn4_w3 = (const float*)d_in[14]; const float* pn4_b3 = (const float*)d_in[15];
    const float* enc_w1 = (const float*)d_in[16]; const float* enc_b1 = (const float*)d_in[17];
    const float* enc_w2 = (const float*)d_in[18]; const float* enc_b2 = (const float*)d_in[19];
    const float* enc_w3 = (const float*)d_in[20]; const float* enc_b3 = (const float*)d_in[21];
    const float* bn1_g  = (const float*)d_in[22]; const float* bn1_b  = (const float*)d_in[23];
    const float* bn2_g  = (const float*)d_in[24]; const float* bn2_b  = (const float*)d_in[25];
    const float* dec_w1 = (const float*)d_in[26]; const float* dec_b1 = (const float*)d_in[27];
    const float* dec_w2 = (const float*)d_in[28]; const float* dec_b2 = (const float*)d_in[29];
    const float* dec_w3 = (const float*)d_in[30]; const float* dec_b3 = (const float*)d_in[31];
    const float* dec_w4 = (const float*)d_in[32]; const float* dec_b4 = (const float*)d_in[33];
    const int*   finit  = (const int*)d_in[34];
    (void)in_sizes; (void)n_in; (void)out_size; (void)ws_size;

    // output layout: refine[Mc], global_feat[Bc*1024], object_scale[Bc], text[Bc*512]
    float* out       = (float*)d_out;
    float* refine    = out;
    float* gfeat     = out + Mc;
    float* scale_out = out + Mc + Bc * 1024;
    float* text_out  = out + Mc + Bc * 1024 + Bc;

    // workspace bump allocator (deterministic)
    size_t off = 0;
    unsigned char* ws = (unsigned char*)d_ws;
    auto alloc = [&](size_t bytes) -> void* {
        void* p = ws + off;
        off = (off + bytes + 255) & ~(size_t)255;
        return p;
    };
    float* xs     = (float*)alloc((size_t)Bc * Nc * 4);
    float* ys     = (float*)alloc((size_t)Bc * Nc * 4);
    float* zs     = (float*)alloc((size_t)Bc * Nc * 4);
    int*   idx    = (int*)alloc((size_t)Mc * 4);
    float* pcn    = (float*)alloc((size_t)Mc * 3 * 4);
    float* enc_in = (float*)alloc((size_t)Mc * 4 * 4);
    _Float16* local3 = (_Float16*)alloc((size_t)Mc * 64 * 2);
    _Float16* local4 = (_Float16*)alloc((size_t)Mc * 64 * 2);
    _Float16* h128a  = (_Float16*)alloc((size_t)Mc * 128 * 2);
    _Float16* h128b  = (_Float16*)alloc((size_t)Mc * 128 * 2);
    float* g4   = (float*)alloc((size_t)Bc * 1024 * 4);
    float* e1   = (float*)alloc((size_t)Bc * 512 * 4);
    float* e2   = (float*)alloc((size_t)Bc * 256 * 4);
    float* hv   = (float*)alloc((size_t)Bc * 128 * 4);
    float* cv64 = (float*)alloc((size_t)Bc * 64 * 4);
    float* cvec = (float*)alloc((size_t)Bc * 512 * 4);
    _Float16* h1  = (_Float16*)alloc((size_t)Mc * 512 * 2);
    _Float16* h2b = (_Float16*)alloc((size_t)Mc * 256 * 2);
    _Float16* h3  = (_Float16*)alloc((size_t)Mc * 128 * 2);
    _Float16* w3w2h = (_Float16*)alloc(64 * 128 * 2);
    _Float16* w3w3h = (_Float16*)alloc(128 * 1024 * 2);
    _Float16* w4w2h = (_Float16*)alloc(64 * 128 * 2);
    _Float16* w4w3h = (_Float16*)alloc(128 * 1024 * 2);
    _Float16* wlh   = (_Float16*)alloc(64 * 512 * 2);
    _Float16* dw2h  = (_Float16*)alloc(512 * 256 * 2);
    _Float16* dw3h  = (_Float16*)alloc(256 * 128 * 2);

    // --- geometry prep ---
    soa_kernel<<<(Bc * Nc + 255) / 256, 256, 0, stream>>>(mesh, xs, ys, zs);
    scale_kernel<<<Bc, 256, 0, stream>>>(xs, ys, zs, scale_out);

    // --- farthest point sampling (one WGP per batch, distance in LDS) ---
    size_t fps_lds = (size_t)Nc * 4 + 1024 * 4 + 1024 * 4 + 16;
    fps_kernel<<<Bc, 1024, fps_lds, stream>>>(xs, ys, zs, finit, idx);

    gather_kernel<<<(Mc + 255) / 256, 256, 0, stream>>>(xs, ys, zs, cmap, idx, scale_out, pcn, enc_in);

    // --- weight conversion to f16 (independent of FPS) ---
    convert_f16_kernel<<<(64 * 128 + 255) / 256, 256, 0, stream>>>(pn3_w2, w3w2h, 64 * 128);
    convert_f16_kernel<<<(128 * 1024 + 255) / 256, 256, 0, stream>>>(pn3_w3, w3w3h, 128 * 1024);
    convert_f16_kernel<<<(64 * 128 + 255) / 256, 256, 0, stream>>>(pn4_w2, w4w2h, 64 * 128);
    convert_f16_kernel<<<(128 * 1024 + 255) / 256, 256, 0, stream>>>(pn4_w3, w4w3h, 128 * 1024);
    convert_f16_kernel<<<(64 * 512 + 255) / 256, 256, 0, stream>>>(dec_w1 + 1024 * 512, wlh, 64 * 512);
    convert_f16_kernel<<<(512 * 256 + 255) / 256, 256, 0, stream>>>(dec_w2, dw2h, 512 * 256);
    convert_f16_kernel<<<(256 * 128 + 255) / 256, 256, 0, stream>>>(dec_w3, dw3h, 256 * 128);

    // --- zero-init max-pool accumulators (relu outputs >= 0) ---
    zero_f32_kernel<<<(Bc * 1024 + 255) / 256, 256, 0, stream>>>(gfeat, Bc * 1024);
    zero_f32_kernel<<<(Bc * 1024 + 255) / 256, 256, 0, stream>>>(g4, Bc * 1024);

    // --- PointNet3 on pcn: 3->64 (VALU), 64->128 (WMMA), 128->1024+maxpool (WMMA) ---
    mlp1_kernel<3><<<(Mc * 64 + 255) / 256, 256, 0, stream>>>(pcn, pn3_w1, pn3_b1, local3);
    wmma_gemm_kernel<1, 0><<<dim3(128 / 64, Mc / 64), 256, 0, stream>>>(
        local3, w3w2h, pn3_b2, nullptr, h128a, nullptr, 64, 128);
    wmma_gemm_kernel<1, 1><<<dim3(1024 / 64, Mc / 64), 256, 0, stream>>>(
        h128a, w3w3h, pn3_b3, nullptr, nullptr, gfeat, 128, 1024);

    // --- PointNet4 on enc_in: 4->64, 64->128, 128->1024+maxpool ---
    mlp1_kernel<4><<<(Mc * 64 + 255) / 256, 256, 0, stream>>>(enc_in, pn4_w1, pn4_b1, local4);
    wmma_gemm_kernel<1, 0><<<dim3(128 / 64, Mc / 64), 256, 0, stream>>>(
        local4, w4w2h, pn4_b2, nullptr, h128b, nullptr, 64, 128);
    wmma_gemm_kernel<1, 1><<<dim3(1024 / 64, Mc / 64), 256, 0, stream>>>(
        h128b, w4w3h, pn4_b3, nullptr, nullptr, g4, 128, 1024);

    // --- VAE encoder over B=16 rows ---
    gemv_kernel<<<(Bc * 512 + 255) / 256, 256, 0, stream>>>(g4, enc_w1, enc_b1, e1, 1024, 512);
    bn_relu_kernel<<<(512 + 255) / 256, 256, 0, stream>>>(e1, bn1_g, bn1_b, 512);
    gemv_kernel<<<(Bc * 256 + 255) / 256, 256, 0, stream>>>(e1, enc_w2, enc_b2, e2, 512, 256);
    bn_relu_kernel<<<(256 + 255) / 256, 256, 0, stream>>>(e2, bn2_g, bn2_b, 256);
    gemv_kernel<<<(Bc * 128 + 255) / 256, 256, 0, stream>>>(e2, enc_w3, enc_b3, hv, 256, 128);
    cvec64_kernel<<<(Bc * 64 + 255) / 256, 256, 0, stream>>>(hv, eps, cv64);

    // --- fold broadcast rows of dec_w1 into per-batch constant cvec ---
    cvec_kernel<<<(Bc * 512 + 255) / 256, 256, 0, stream>>>(
        gfeat, text, scale_out, cv64, dec_w1, dec_b1, cvec);

    // --- decoder: leaky GEMM chain (per-point part only: local3 @ W_l + cvec) ---
    wmma_gemm_kernel<2, 0><<<dim3(512 / 64, Mc / 64), 256, 0, stream>>>(
        local3, wlh, nullptr, cvec, h1, nullptr, 64, 512);
    wmma_gemm_kernel<2, 0><<<dim3(256 / 64, Mc / 64), 256, 0, stream>>>(
        h1, dw2h, dec_b2, nullptr, h2b, nullptr, 512, 256);
    wmma_gemm_kernel<2, 0><<<dim3(128 / 64, Mc / 64), 256, 0, stream>>>(
        h2b, dw3h, dec_b3, nullptr, h3, nullptr, 256, 128);

    refine_kernel<<<(Mc + 255) / 256, 256, 0, stream>>>(h3, dec_w4, dec_b4, refine);
    copy_f32_kernel<<<(Bc * 512 + 255) / 256, 256, 0, stream>>>(text, text_out, Bc * 512);
}
